// SpiralAttentionMixer_74577812127883
// MI455X (gfx1250) — matrix operand order repaired and checked by reading, not compiled, not run on hardware
//
#include <hip/hip_runtime.h>
#include <hip/hip_bf16.h>

// ---------------------------------------------------------------------------
// Spiral attention mixer for MI455X (gfx1250), bf16 WMMA pipeline.
// Stages: fp32->bf16 convert | QK/V projections (WMMA GEMM, V stored
// transposed [B,H,D,T]) | flash-style masked attention (WMMA S and PV,
// online softmax, LDS P-staging, D split across wave pairs) | output
// projection (WMMA GEMM, f32 out).  Wave-uniform control state is forced
// into SGPRs via readfirstlane so loops are scalar and EXEC stays all-ones.
// ---------------------------------------------------------------------------

typedef __attribute__((ext_vector_type(16))) __bf16 bf16x16;
typedef __attribute__((ext_vector_type(8)))  __bf16 bf16x8;
typedef __attribute__((ext_vector_type(8)))  float  f32x8;

constexpr int   kB = 2, kT = 2048, kC = 1024, kH = 4, kD = 256;
constexpr int   kM = kB * kT;            // 4096 token rows
constexpr float kScale = 0.0625f;        // 1/sqrt(D) = 1/16

// D = A(16x32) x B(32x16) + C, bf16 inputs, f32 accumulate
__device__ __forceinline__ f32x8 wmma_bf16(bf16x16 a, bf16x16 b, f32x8 c) {
  return __builtin_amdgcn_wmma_f32_16x16x32_bf16(
      /*neg_a=*/false, a, /*neg_b=*/false, b,
      /*c_mod=*/(short)0, c, /*reuse_a=*/false, /*reuse_b=*/false);
}

// A-fragment (16x32, MxK) from row-major [16, ld] bf16 tile.
// ISA layout: lanes 0-15 (g=0): K = {0..7, 16..23}; lanes 16-31 (g=1):
// K = {8..15, 24..31}. Two contiguous 16-byte loads per lane.
__device__ __forceinline__ bf16x16 load_fragA(const __bf16* tile, int ld) {
  const int lane = threadIdx.x & 31;
  const int m = lane & 15, g = lane >> 4;
  const __bf16* p = tile + (size_t)m * ld + 8 * g;
  bf16x8 lo = *reinterpret_cast<const bf16x8*>(p);
  bf16x8 hi = *reinterpret_cast<const bf16x8*>(p + 16);
  bf16x16 f;
#pragma unroll
  for (int i = 0; i < 8; ++i) { f[i] = lo[i]; f[i + 8] = hi[i]; }
  return f;
}

// B-fragment (32x16, KxN) from row-major [16, ld] tile holding B^T
// (operand stored as [N rows, K cols] -- the NT-GEMM layout).
// ISA layout: lane n holds column n; lanes 0-15: K = {0..7, 16..23}, lanes
// 16-31: K = {8..15, 24..31}. Two contiguous 16-byte loads per lane.
__device__ __forceinline__ bf16x16 load_fragB(const __bf16* tile, int ld) {
  const int lane = threadIdx.x & 31;
  const int n = lane & 15, g = lane >> 4;
  const __bf16* p = tile + (size_t)n * ld + 8 * g;
  bf16x8 lo = *reinterpret_cast<const bf16x8*>(p);
  bf16x8 hi = *reinterpret_cast<const bf16x8*>(p + 16);
  bf16x16 f;
#pragma unroll
  for (int i = 0; i < 8; ++i) { f[i] = lo[i]; f[i + 8] = hi[i]; }
  return f;
}

__global__ void cvt_f32_to_bf16_kernel(const float* __restrict__ in,
                                       __bf16* __restrict__ out, int n) {
  int i = blockIdx.x * blockDim.x + threadIdx.x;
  if (i < n) out[i] = (__bf16)in[i];
}

// ---------------------------------------------------------------------------
// NT GEMM: out[m, n] = sum_k A[m,k] * W[n,k] + bias[n].
// M = kM = 4096, N = K = kC = 1024 (all stages share these dims).
// One wave -> 16 rows x 64 cols (4 C-tiles share one A-fragment per k-step).
// MODE 0: bf16 row-major out. MODE 1: f32 row-major out.
// MODE 2: bf16 out scattered to transposed-V layout [B,H,D,T].
// ---------------------------------------------------------------------------
template <int MODE>
__global__ void gemm_nt_kernel(const __bf16* __restrict__ A,
                               const __bf16* __restrict__ W,
                               const float*  __restrict__ bias,
                               void* __restrict__ out) {
  constexpr int K = kC, N = kC;
  // wave id is identical across the 32 lanes: pin it (and everything derived
  // from it) into SGPRs so addressing/loop control is scalar.
  const int wave = __builtin_amdgcn_readfirstlane(
      blockIdx.x * (blockDim.x >> 5) + (threadIdx.x >> 5));
  const int tm  = wave >> 4;          // 16 groups of 64 columns across N
  const int tn0 = (wave & 15) << 2;   // first 16-col tile of this 64-col group

  const __bf16* arow = A + (size_t)tm * 16 * K;
  const __bf16* wrow[4];
#pragma unroll
  for (int t = 0; t < 4; ++t) wrow[t] = W + (size_t)(tn0 + t) * 16 * K;

  f32x8 acc[4];
#pragma unroll
  for (int t = 0; t < 4; ++t)
#pragma unroll
    for (int r = 0; r < 8; ++r) acc[t][r] = 0.f;

  // Software-pipelined K loop: fetch k+32 fragments while k's WMMAs issue.
  bf16x16 af = load_fragA(arow, K);
  bf16x16 bf[4];
#pragma unroll
  for (int t = 0; t < 4; ++t) bf[t] = load_fragB(wrow[t], K);
  for (int k = 0; k < K; k += 32) {
    bf16x16 afn = af;
    bf16x16 bfn[4];
#pragma unroll
    for (int t = 0; t < 4; ++t) bfn[t] = bf[t];
    if (k + 32 < K) {
      af = load_fragA(arow + k + 32, K);
#pragma unroll
      for (int t = 0; t < 4; ++t) bf[t] = load_fragB(wrow[t] + k + 32, K);
    }
#pragma unroll
    for (int t = 0; t < 4; ++t) acc[t] = wmma_bf16(afn, bfn[t], acc[t]);
  }

  const int lane = threadIdx.x & 31;
  const int n = lane & 15, hsel = lane >> 4;
#pragma unroll
  for (int t = 0; t < 4; ++t) {
    const int cn = (tn0 + t) * 16 + n;
    const float bn = bias[cn];
#pragma unroll
    for (int r = 0; r < 8; ++r) {
      const int rm = tm * 16 + r + 8 * hsel;
      const float val = acc[t][r] + bn;
      if constexpr (MODE == 1) {
        reinterpret_cast<float*>(out)[(size_t)rm * N + cn] = val;
      } else if constexpr (MODE == 0) {
        reinterpret_cast<__bf16*>(out)[(size_t)rm * N + cn] = (__bf16)val;
      } else {  // MODE == 2: Vt[b][h][d][t]
        const int bb = rm >> 11, tt = rm & (kT - 1);
        const int hh = cn >> 8,  dd = cn & (kD - 1);
        reinterpret_cast<__bf16*>(out)
            [((size_t)(bb * kH + hh) * kD + dd) * kT + tt] = (__bf16)val;
      }
    }
  }
}

// Spiral mask: p allowed for row i iff p <= i and (p-i) mod T lands on the
// stride lattice: with u = (p-i) & (T-1), v = (u < T/2) ? u+T/2 : u-T/2,
// allowed iff v % stride == 0.  (stride 1 => plain causal.)
__device__ __forceinline__ bool spiral_ok(int i, int p, int s) {
  if (p > i) return false;
  const int u = (p - i) & (kT - 1);
  const int v = (u < kT / 2) ? (u + kT / 2) : (u - kT / 2);
  return (v % s) == 0;
}

// ---------------------------------------------------------------------------
// Flash-style spiral attention. qk: [B*T, C] bf16 (q == k). vt: [B,H,D,T]
// bf16. ao: [B*T, C] bf16 (head-concatenated attention output).
// One wave owns a 16-row i-tile x 128-wide D-half of one (b,h): the D split
// keeps the O accumulator at 8 C-tiles (64 VGPRs) so nothing spills. Both
// waves of a pair recompute S (scores are the cheap half here vs spilling).
// ---------------------------------------------------------------------------
__global__ void __attribute__((amdgpu_waves_per_eu(1)))
spiral_attn_kernel(const __bf16* __restrict__ qk,
                   const __bf16* __restrict__ vt,
                   __bf16* __restrict__ ao) {
  __shared__ __align__(16) __bf16 ldsP[8][16 * 32];  // per-wave P staging

  const int wid  = threadIdx.x >> 5;
  const int lane = threadIdx.x & 31;
  const int wave = __builtin_amdgcn_readfirstlane(
      blockIdx.x * (blockDim.x >> 5) + wid);
  const int dhalf = wave & 1;               // which 128-wide half of D
  const int itile = (wave >> 1) & 127;      // T/16 = 128 row tiles
  const int h     = (wave >> 8) & (kH - 1);
  const int b     = wave >> 10;
  const int row0  = itile * 16;
  const int s = (h == 0) ? 1 : (h == 1) ? 3 : (h == 2) ? 7 : 13;

  const __bf16* qbase = qk + ((size_t)b * kT + row0) * kC + h * kD;
  const __bf16* kbase = qk + (size_t)b * kT * kC + h * kD;
  const __bf16* vbase = vt + ((size_t)b * kH + h) * (size_t)kD * kT +
                        (size_t)dhalf * 8 * 16 * kT;

  const int nl   = lane & 15;
  const int hsel = lane >> 4;

  f32x8 O[8];
#pragma unroll
  for (int dt = 0; dt < 8; ++dt)
#pragma unroll
    for (int r = 0; r < 8; ++r) O[dt][r] = 0.f;
  float Mr[8], Lr[8];
#pragma unroll
  for (int r = 0; r < 8; ++r) { Mr[r] = -1e30f; Lr[r] = 0.f; }

  const int imax = row0 + 15;                  // scalar loop bound
  for (int col0 = 0; col0 <= imax; col0 += 32) {
    // Warm L2/WGP$ for the next column block (global_prefetch_b8 path).
    if (col0 + 32 <= imax) {
      __builtin_prefetch(kbase + (size_t)(col0 + 32) * kC, 0, 3);
      __builtin_prefetch(vbase + (col0 + 32), 0, 3);
    }

    // ---- S = Q . K^T over two 16-col tiles (K-dim = D = 256) ----
    f32x8 s0 = {}, s1 = {};
    const __bf16* k0 = kbase + (size_t)col0 * kC;
    const __bf16* k1 = k0 + (size_t)16 * kC;
    bf16x16 qa = load_fragA(qbase, kC);
    bf16x16 kb0 = load_fragB(k0, kC);
    bf16x16 kb1 = load_fragB(k1, kC);
#pragma unroll
    for (int k = 0; k < kD; k += 32) {
      bf16x16 qan = qa, kb0n = kb0, kb1n = kb1;
      if (k + 32 < kD) {                       // fetch ahead of the WMMAs
        qa  = load_fragA(qbase + k + 32, kC);
        kb0 = load_fragB(k0 + k + 32, kC);
        kb1 = load_fragB(k1 + k + 32, kC);
      }
      s0 = wmma_bf16(qan, kb0n, s0);
      s1 = wmma_bf16(qan, kb1n, s1);
    }

    // ---- mask + scale folded in place, then block row-max ----
    float tmax[8];
#pragma unroll
    for (int r = 0; r < 8; ++r) {
      const int i  = row0 + r + 8 * hsel;
      s0[r] = spiral_ok(i, col0 + nl, s)      ? s0[r] * kScale : -1e30f;
      s1[r] = spiral_ok(i, col0 + 16 + nl, s) ? s1[r] * kScale : -1e30f;
      tmax[r] = fmaxf(s0[r], s1[r]);
    }
#pragma unroll
    for (int m = 1; m < 16; m <<= 1)
#pragma unroll
      for (int r = 0; r < 8; ++r)
        tmax[r] = fmaxf(tmax[r], __shfl_xor(tmax[r], m, 32));

    // ---- online softmax update ----
    float corr[8];
#pragma unroll
    for (int r = 0; r < 8; ++r) {
      const float nm = fmaxf(Mr[r], tmax[r]);
      corr[r] = __expf(Mr[r] - nm);
      Mr[r]   = nm;
    }
    float rsum[8];
#pragma unroll
    for (int r = 0; r < 8; ++r) {
      const float e0 = (s0[r] > -1e29f) ? __expf(s0[r] - Mr[r]) : 0.f;
      const float e1 = (s1[r] > -1e29f) ? __expf(s1[r] - Mr[r]) : 0.f;
      const int row = r + 8 * hsel;
      ldsP[wid][row * 32 + nl]      = (__bf16)e0;   // C-layout -> row-major P
      ldsP[wid][row * 32 + 16 + nl] = (__bf16)e1;
      rsum[r] = e0 + e1;
    }
#pragma unroll
    for (int m = 1; m < 16; m <<= 1)
#pragma unroll
      for (int r = 0; r < 8; ++r)
        rsum[r] += __shfl_xor(rsum[r], m, 32);
#pragma unroll
    for (int r = 0; r < 8; ++r) Lr[r] = Lr[r] * corr[r] + rsum[r];
#pragma unroll
    for (int dt = 0; dt < 8; ++dt)
#pragma unroll
      for (int r = 0; r < 8; ++r) O[dt][r] *= corr[r];

    // same-wave LDS store->load hazard: drain DScnt, then pick up the P
    // tile as a 16x32 bf16 A-fragment (no cross-wave barrier needed).
    asm volatile("s_wait_dscnt 0" ::: "memory");
    const bf16x16 pf = load_fragA(&ldsP[wid][0], 32);

    // ---- O += P . V for this wave's 128-wide D-half (double-buffered) ----
    const __bf16* vcol = vbase + col0;
    bf16x16 vb = load_fragB(vcol, kT);
#pragma unroll
    for (int dt = 0; dt < 8; ++dt) {
      bf16x16 vnext = vb;
      if (dt < 7) vnext = load_fragB(vcol + (size_t)(dt + 1) * 16 * kT, kT);
      O[dt] = wmma_bf16(pf, vb, O[dt]);
      vb = vnext;
    }
  }

  // ---- epilogue: normalize (rows with empty mask -> zeros) and store ----
  float inv[8];
#pragma unroll
  for (int r = 0; r < 8; ++r) inv[r] = (Lr[r] > 0.f) ? 1.f / Lr[r] : 0.f;
#pragma unroll
  for (int dt = 0; dt < 8; ++dt)
#pragma unroll
    for (int r = 0; r < 8; ++r) {
      const int i = row0 + r + 8 * hsel;
      const int d = (dhalf * 8 + dt) * 16 + nl;
      ao[((size_t)b * kT + i) * kC + h * kD + d] = (__bf16)(O[dt][r] * inv[r]);
    }
}

// ---------------------------------------------------------------------------
extern "C" void kernel_launch(void* const* d_in, const int* in_sizes, int n_in,
                              void* d_out, int out_size, void* d_ws, size_t ws_size,
                              hipStream_t stream) {
  (void)in_sizes; (void)n_in; (void)out_size; (void)ws_size;
  const float* x   = (const float*)d_in[0];
  const float* Wqk = (const float*)d_in[1];
  const float* bqk = (const float*)d_in[2];
  const float* Wv  = (const float*)d_in[3];
  const float* bv  = (const float*)d_in[4];
  const float* Wo  = (const float*)d_in[5];
  const float* bo  = (const float*)d_in[6];

  char* ws = (char*)d_ws;
  size_t off = 0;
  auto alloc_bf16 = [&](size_t elems) {
    __bf16* p = (__bf16*)(ws + off);
    off += elems * sizeof(__bf16);
    off = (off + 255) & ~(size_t)255;
    return p;
  };
  __bf16* xb   = alloc_bf16((size_t)kM * kC);  // x in bf16
  __bf16* wqkb = alloc_bf16((size_t)kC * kC);
  __bf16* wvb  = alloc_bf16((size_t)kC * kC);
  __bf16* wob  = alloc_bf16((size_t)kC * kC);
  __bf16* qkb  = alloc_bf16((size_t)kM * kC);  // q (== k) projections
  __bf16* vtb  = alloc_bf16((size_t)kM * kC);  // V transposed [B,H,D,T]
  __bf16* aob  = alloc_bf16((size_t)kM * kC);  // attention output [B*T, C]

  const int nx = kM * kC, nw = kC * kC;
  cvt_f32_to_bf16_kernel<<<(nx + 255) / 256, 256, 0, stream>>>(x,   xb,   nx);
  cvt_f32_to_bf16_kernel<<<(nw + 255) / 256, 256, 0, stream>>>(Wqk, wqkb, nw);
  cvt_f32_to_bf16_kernel<<<(nw + 255) / 256, 256, 0, stream>>>(Wv,  wvb,  nw);
  cvt_f32_to_bf16_kernel<<<(nw + 255) / 256, 256, 0, stream>>>(Wo,  wob,  nw);

  // (M/16)*(N/64) waves, 8 waves per 256-thread block -> 512 blocks.
  const int gemm_blocks = (kM / 16) * (kC / 64) / 8;
  gemm_nt_kernel<0><<<gemm_blocks, 256, 0, stream>>>(xb, wqkb, bqk, qkb);
  gemm_nt_kernel<2><<<gemm_blocks, 256, 0, stream>>>(xb, wvb,  bv,  vtb);

  // B*H*(T/16)*2 = 2048 waves -> 256 blocks of 8 waves.
  spiral_attn_kernel<<<kB * kH * (kT / 16) * 2 / 8, 256, 0, stream>>>(qkb, vtb, aob);

  gemm_nt_kernel<1><<<gemm_blocks, 256, 0, stream>>>(aob, wob, bo, (float*)d_out);
}
